// FraudDetector_14628658610327
// MI455X (gfx1250) — matrix operand order repaired and checked by reading, not compiled, not run on hardware
//
#include <hip/hip_runtime.h>
#include <math.h>

typedef __attribute__((ext_vector_type(16))) _Float16 v16h;
typedef __attribute__((ext_vector_type(8)))  float    v8f;

#define CDIV(a,b) ((int)((((long)(a)) + (b) - 1) / (b)))

// ---------- order-preserving float<->uint encoding for atomicMax segment-max ----------
__device__ __forceinline__ unsigned enc_f32(float f) {
    unsigned u = __float_as_uint(f);
    return (u & 0x80000000u) ? ~u : (u | 0x80000000u);
}
__device__ __forceinline__ float dec_f32(unsigned e) {
    unsigned u = (e & 0x80000000u) ? (e ^ 0x80000000u) : ~e;
    return __uint_as_float(u);
}
#define ENC_NEG_INF 0x007FFFFFu   // enc(-inf)

// ---------- pack W (f32, K x N row-major) into WMMA B-fragment f16 layout ----------
// out[((kt*tilesN + tn)*32 + lane)*16 + i] = (f16) W[kt*32 + (lane>>4)*16 + i][tn*16 + (lane&15)]
__global__ void pack_b_f16(const float* __restrict__ W, int N, long total,
                           _Float16* __restrict__ out) {
    long t = blockIdx.x * (long)blockDim.x + threadIdx.x;
    if (t >= total) return;
    int  i      = (int)(t & 15);
    int  lane   = (int)((t >> 4) & 31);
    long grp    = t >> 9;                 // kt*tilesN + tn
    int  tilesN = N >> 4;
    long kt     = grp / tilesN;
    int  tn     = (int)(grp - kt * tilesN);
    int  k      = (int)(kt * 32 + ((lane >> 4) * 16) + i);
    int  c      = tn * 16 + (lane & 15);
    out[t] = (_Float16)W[(size_t)k * N + c];
}

// ---------- WMMA GEMM: C[M,N] = A[M,K] @ W[K,N] (+bias); A f32 on the fly, B pre-packed f16.
// One wave computes a 16 x (NT*16) strip: A fragment reused for NT WMMAs per k-step.
// Requirements: M%16==0, (N/16)%NT==0, K%32==0, lda%4==0.
template <int NT>
__global__ void __launch_bounds__(128)
gemm_wmma_packed(const float* __restrict__ A, int lda,
                 const _Float16* __restrict__ Bp,
                 const float* __restrict__ bias,
                 float* __restrict__ C, int ldc,
                 int M, int N, int K)
{
    const int lane   = threadIdx.x & 31;
    const int wid    = blockIdx.x * (blockDim.x >> 5) + (threadIdx.x >> 5);
    const int tilesN = N >> 4;
    const int gN     = tilesN / NT;
    const int groups = (M >> 4) * gN;
    if (wid >= groups) return;
    const int tm   = wid / gN;
    const int tg   = wid - tm * gN;
    const int l16  = lane & 15;
    const int hi   = lane >> 4;
    const int arow = tm * 16 + l16;

    v8f acc[NT];
#pragma unroll
    for (int nt = 0; nt < NT; ++nt)
#pragma unroll
        for (int r = 0; r < 8; ++r) acc[nt][r] = 0.f;

    const int nk = K >> 5;
    for (int kt = 0; kt < nk; ++kt) {
        // A fragment (16x32 f16): element i -> K = kt*32 + hi*16 + i
        v16h a;
        const float* Ap = A + (size_t)arow * lda + kt * 32 + hi * 16;
#pragma unroll
        for (int g = 0; g < 2; ++g) {
            float4 u0 = *(const float4*)(Ap + g * 8 + 0);
            float4 u1 = *(const float4*)(Ap + g * 8 + 4);
            a[g*8+0] = (_Float16)u0.x; a[g*8+1] = (_Float16)u0.y;
            a[g*8+2] = (_Float16)u0.z; a[g*8+3] = (_Float16)u0.w;
            a[g*8+4] = (_Float16)u1.x; a[g*8+5] = (_Float16)u1.y;
            a[g*8+6] = (_Float16)u1.z; a[g*8+7] = (_Float16)u1.w;
        }
        if (kt + 1 < nk) __builtin_prefetch(Ap + 32, 0, 1);   // next A k-slice

        const _Float16* bbase = Bp + (((size_t)kt * tilesN + tg * NT) * 32 + lane) * 16;
#pragma unroll
        for (int nt = 0; nt < NT; ++nt) {
            v16h b = *(const v16h*)(bbase + (size_t)nt * 512);
            acc[nt] = __builtin_amdgcn_wmma_f32_16x16x32_f16(
                          false, a, false, b, (short)0, acc[nt], false, false);
        }
    }
#pragma unroll
    for (int nt = 0; nt < NT; ++nt) {
        const int bcol = (tg * NT + nt) * 16 + l16;
        const float bv = bias ? bias[bcol] : 0.0f;
#pragma unroll
        for (int r = 0; r < 8; ++r)   // C/D: VGPR r -> M = tm*16 + hi*8 + r
            C[(size_t)(tm * 16 + hi * 8 + r) * ldc + bcol] = acc[nt][r] + bv;
    }
}

// ---------- small helper kernels ----------
__global__ void fill_f32(float* __restrict__ p, float v, long n) {
    long i = blockIdx.x * (long)blockDim.x + threadIdx.x;
    if (i < n) p[i] = v;
}
__global__ void fill_u32(unsigned* __restrict__ p, unsigned v, long n) {
    long i = blockIdx.x * (long)blockDim.x + threadIdx.x;
    if (i < n) p[i] = v;
}

// h_post[n,c] += sum_{j<5} meta[n,j] * W[(768+j), c]   (rank-5 tail of the 773-K GEMM)
__global__ void add_meta(const float* __restrict__ meta, const float* __restrict__ W,
                         float* __restrict__ hp, long n) {
    long t = blockIdx.x * (long)blockDim.x + threadIdx.x;
    if (t >= n) return;
    long nd = t >> 6; int c = (int)(t & 63);
    float s = 0.f;
#pragma unroll
    for (int j = 0; j < 5; ++j) s += meta[nd * 5 + j] * W[(768 + j) * 64 + c];
    hp[t] += s;
}

// Fold Wd (64x128) with ad (2x64) into Amat[64][2]: Amat[k][h] = sum_c Wd[k, h*64+c]*ad[h,c]
__global__ void make_alpha_mat(const float* __restrict__ Wd, const float* __restrict__ av,
                               float* __restrict__ Amat) {
    int t = threadIdx.x;            // 0..127
    int k = t >> 1, hh = t & 1;
    float s = 0.f;
    for (int c = 0; c < 64; ++c) s += Wd[k * 128 + hh * 64 + c] * av[hh * 64 + c];
    Amat[k * 2 + hh] = s;
}

// a_d[n,h] = h_dst[n,:] @ Amat[:,h]
__global__ void node_alpha(const float* __restrict__ h, const float* __restrict__ Amat,
                           float* __restrict__ out, int Nn) {
    int n = blockIdx.x * blockDim.x + threadIdx.x;
    if (n >= Nn) return;
    float s0 = 0.f, s1 = 0.f;
    for (int k = 0; k < 64; ++k) {
        float v = h[(size_t)n * 64 + k];
        s0 += v * Amat[k * 2]; s1 += v * Amat[k * 2 + 1];
    }
    out[n * 2] = s0; out[n * 2 + 1] = s1;
}

// a_s[n,h] = sum_c hs[n, h*64+c] * as[h,c]
__global__ void alpha_from_hs(const float* __restrict__ hs, const float* __restrict__ av,
                              float* __restrict__ out, int Nn) {
    int n = blockIdx.x * blockDim.x + threadIdx.x;
    if (n >= Nn) return;
    float s0 = 0.f, s1 = 0.f;
    for (int c = 0; c < 64; ++c) {
        s0 += hs[(size_t)n * 128 + c]      * av[c];
        s1 += hs[(size_t)n * 128 + 64 + c] * av[64 + c];
    }
    out[n * 2] = s0; out[n * 2 + 1] = s1;
}

// per-edge logits + segment max (encoded atomicMax)
__global__ void edge_logits(const int* __restrict__ src, const int* __restrict__ dst, int E,
                            const float* __restrict__ as_, const float* __restrict__ ad_,
                            float* __restrict__ ebuf, unsigned* __restrict__ menc) {
    int e = blockIdx.x * blockDim.x + threadIdx.x;
    if (e >= E) return;
    int s = src[e], d = dst[e];
#pragma unroll
    for (int hh = 0; hh < 2; ++hh) {
        float x = as_[s * 2 + hh] + ad_[d * 2 + hh];
        x = x > 0.f ? x : 0.2f * x;                  // leaky_relu(x, 0.2)
        ebuf[e * 2 + hh] = x;
        atomicMax(&menc[d * 2 + hh], enc_f32(x));
    }
}

// decode segment max in place; empty segments (-inf) -> 0
__global__ void decode_m(unsigned* __restrict__ m, long n) {
    long i = blockIdx.x * (long)blockDim.x + threadIdx.x;
    if (i >= n) return;
    float f = dec_f32(m[i]);
    if (f < -3.0e38f) f = 0.f;
    ((float*)m)[i] = f;
}

// w = exp(e - m[dst]); segment sum of w
__global__ void edge_exp(const int* __restrict__ dst, int E, float* __restrict__ ebuf,
                         const float* __restrict__ m, float* __restrict__ den) {
    int e = blockIdx.x * blockDim.x + threadIdx.x;
    if (e >= E) return;
    int d = dst[e];
#pragma unroll
    for (int hh = 0; hh < 2; ++hh) {
        float w = expf(ebuf[e * 2 + hh] - m[d * 2 + hh]);
        ebuf[e * 2 + hh] = w;
        atomicAdd(&den[d * 2 + hh], w);
    }
}

// agg[dst,c] += 0.5*(hs[src,0,c]*coef0 + hs[src,1,c]*coef1); 4 channels per thread
__global__ void edge_scatter(const int* __restrict__ src, const int* __restrict__ dst, long E,
                             const float* __restrict__ w, const float* __restrict__ den,
                             const float* __restrict__ hs, float* __restrict__ agg) {
    long t = blockIdx.x * (long)blockDim.x + threadIdx.x;
    if (t >= E * 16) return;
    long e = t >> 4; int c4 = (int)(t & 15) * 4;
    int s = src[e], d = dst[e];
    float c0 = w[e * 2]     / (den[d * 2]     + 1e-16f);
    float c1 = w[e * 2 + 1] / (den[d * 2 + 1] + 1e-16f);
    float4 h0 = *(const float4*)(hs + (size_t)s * 128 + c4);
    float4 h1 = *(const float4*)(hs + (size_t)s * 128 + 64 + c4);
    float* ap = agg + (size_t)d * 64 + c4;
    atomicAdd(ap + 0, 0.5f * (h0.x * c0 + h1.x * c1));
    atomicAdd(ap + 1, 0.5f * (h0.y * c0 + h1.y * c1));
    atomicAdd(ap + 2, 0.5f * (h0.z * c0 + h1.z * c1));
    atomicAdd(ap + 3, 0.5f * (h0.w * c0 + h1.w * c1));
}

__global__ void add_bias_rows(float* __restrict__ agg, const float* __restrict__ b, long n) {
    long t = blockIdx.x * (long)blockDim.x + threadIdx.x;
    if (t < n) agg[t] += b[t & 63];
}

// h = relu(agg) + h
__global__ void layer_update(float* __restrict__ h, const float* __restrict__ agg, long n) {
    long t = blockIdx.x * (long)blockDim.x + threadIdx.x;
    if (t < n) h[t] = fmaxf(agg[t], 0.f) + h[t];
}

// out[n,:] = relu(z[n,:]) @ W2(32x2) + b2
__global__ void cls2k(const float* __restrict__ z, const float* __restrict__ W2,
                      const float* __restrict__ b2, float* __restrict__ out, int n) {
    int i = blockIdx.x * blockDim.x + threadIdx.x;
    if (i >= n) return;
    float s0 = b2[0], s1 = b2[1];
    for (int k = 0; k < 32; ++k) {
        float v = fmaxf(z[(size_t)i * 32 + k], 0.f);
        s0 += v * W2[k * 2]; s1 += v * W2[k * 2 + 1];
    }
    out[i * 2] = s0; out[i * 2 + 1] = s1;
}

// =====================================================================================
extern "C" void kernel_launch(void* const* d_in, const int* in_sizes, int n_in,
                              void* d_out, int out_size, void* d_ws, size_t ws_size,
                              hipStream_t stream)
{
    (void)in_sizes; (void)n_in; (void)out_size; (void)ws_size;
    const int NP = 50000, NU = 100000, NEn = 30000;

    const float* x_post = (const float*)d_in[0];
    const float* meta   = (const float*)d_in[1];
    const float* x_user = (const float*)d_in[2];
    const float* x_ent  = (const float*)d_in[3];
    // edges: d_in[4..9], each [2,E] int32 (src row then dst row)
    const float* pW = (const float*)d_in[10]; const float* pb = (const float*)d_in[11];
    const float* uW = (const float*)d_in[12]; const float* ub = (const float*)d_in[13];
    const float* eW = (const float*)d_in[14]; const float* eb = (const float*)d_in[15];
    const float* c1W = (const float*)d_in[76]; const float* c1b = (const float*)d_in[77];
    const float* c2W = (const float*)d_in[78]; const float* c2b = (const float*)d_in[79];

    // ---- workspace carve-up (256B aligned) ----
    char* wsb = (char*)d_ws;
    size_t off = 0;
    auto carve = [&](size_t bytes) -> char* {
        char* p = wsb + off;
        off = (off + bytes + 255) & ~(size_t)255;
        return p;
    };
    const int nn[3] = { NP, NU, NEn };         // 0=post 1=user 2=entity
    float* h[3];   for (int t = 0; t < 3; ++t) h[t]   = (float*)carve((size_t)nn[t] * 64 * 4);
    float* agg[3]; for (int t = 0; t < 3; ++t) agg[t] = (float*)carve((size_t)nn[t] * 64 * 4);
    float*    hs    = (float*)carve((size_t)NU * 128 * 4);   // max Ns = NU
    float*    asb   = (float*)carve((size_t)NU * 2 * 4);
    float*    adb   = (float*)carve((size_t)NU * 2 * 4);
    unsigned* menc  = (unsigned*)carve((size_t)NU * 2 * 4);  // max-enc, decoded in place
    float*    den   = (float*)carve((size_t)NU * 2 * 4);
    float*    ebuf  = (float*)carve((size_t)400000 * 2 * 4); // max E
    float*    AdMat = (float*)carve(64 * 2 * 4);
    float*    z     = (float*)carve((size_t)NP * 32 * 4);
    // packed f16 weight buffers
    _Float16* pWp = (_Float16*)carve((size_t)768 * 64 * 2);
    _Float16* uWp = (_Float16*)carve((size_t)32 * 64 * 2);
    _Float16* eWp = (_Float16*)carve((size_t)64 * 64 * 2);
    _Float16* c1Wp = (_Float16*)carve((size_t)64 * 32 * 2);
    _Float16* WsP[2][6];
    for (int L = 0; L < 2; ++L)
        for (int r = 0; r < 6; ++r) WsP[L][r] = (_Float16*)carve((size_t)64 * 128 * 2);

    auto pack = [&](const float* W, int K, int N, _Float16* out) {
        long total = (long)(K >> 5) * (N >> 4) * 512;
        pack_b_f16<<<CDIV(total, 256), 256, 0, stream>>>(W, N, total, out);
    };
    auto gemm4 = [&](const float* A, int lda, const _Float16* Bp, const float* bias,
                     float* C, int ldc, int M, int N, int K) {
        int groups = (M >> 4) * ((N >> 4) / 4);
        gemm_wmma_packed<4><<<CDIV(groups, 4), 128, 0, stream>>>(A, lda, Bp, bias, C, ldc, M, N, K);
    };

    // ---- pack all weights into WMMA B layout (f16) ----
    pack(pW, 768, 64, pWp);          // uses rows 0..767 of the 773-row matrix
    pack(uW, 32, 64, uWp);
    pack(eW, 64, 64, eWp);
    pack(c1W, 64, 32, c1Wp);
    for (int L = 0; L < 2; ++L)
        for (int r = 0; r < 6; ++r)
            pack((const float*)d_in[16 + L * 30 + r * 5], 64, 128, WsP[L][r]);

    // ---- input projections ----
    gemm4(x_post, 768, pWp, pb, h[0], 64, NP, 64, 768);                          // K=768 part
    add_meta<<<CDIV((long)NP * 64, 256), 256, 0, stream>>>(meta, pW, h[0], (long)NP * 64);
    gemm4(x_user, 32, uWp, ub, h[1], 64, NU, 64, 32);
    gemm4(x_ent,  64, eWp, eb, h[2], 64, NEn, 64, 64);

    // ---- 2 hetero-GAT layers ----
    const int relE[6] = { 50000, 250000, 150000, 250000, 400000, 150000 };
    const int relS[6] = { 1, 1, 0, 1, 1, 0 };   // src type
    const int relD[6] = { 0, 0, 2, 1, 1, 0 };   // dst type
    for (int L = 0; L < 2; ++L) {
        for (int t = 0; t < 3; ++t)
            fill_f32<<<CDIV((long)nn[t] * 64, 256), 256, 0, stream>>>(agg[t], 0.f, (long)nn[t] * 64);
        for (int r = 0; r < 6; ++r) {
            const int base = 16 + L * 30 + r * 5;
            const float* Wd  = (const float*)d_in[base + 1];
            const float* av  = (const float*)d_in[base + 2];
            const float* adv = (const float*)d_in[base + 3];
            const float* bb  = (const float*)d_in[base + 4];
            const int*   ei  = (const int*)d_in[4 + r];
            const int E = relE[r], st = relS[r], dt = relD[r];
            const int Ns = nn[st], Nd = nn[dt];
            const int* srcI = ei;
            const int* dstI = ei + E;

            gemm4(h[st], 64, WsP[L][r], nullptr, hs, 128, Ns, 128, 64);          // hs = h_src@Ws
            alpha_from_hs<<<CDIV(Ns, 256), 256, 0, stream>>>(hs, av, asb, Ns);
            make_alpha_mat<<<1, 128, 0, stream>>>(Wd, adv, AdMat);               // fold Wd·ad
            node_alpha<<<CDIV(Nd, 256), 256, 0, stream>>>(h[dt], AdMat, adb, Nd);

            fill_u32<<<CDIV((long)Nd * 2, 256), 256, 0, stream>>>(menc, ENC_NEG_INF, (long)Nd * 2);
            fill_f32<<<CDIV((long)Nd * 2, 256), 256, 0, stream>>>(den, 0.f, (long)Nd * 2);
            edge_logits<<<CDIV(E, 256), 256, 0, stream>>>(srcI, dstI, E, asb, adb, ebuf, menc);
            decode_m<<<CDIV((long)Nd * 2, 256), 256, 0, stream>>>(menc, (long)Nd * 2);
            edge_exp<<<CDIV(E, 256), 256, 0, stream>>>(dstI, E, ebuf, (const float*)menc, den);
            edge_scatter<<<CDIV((long)E * 16, 256), 256, 0, stream>>>(srcI, dstI, (long)E,
                                                                      ebuf, den, hs, agg[dt]);
            add_bias_rows<<<CDIV((long)Nd * 64, 256), 256, 0, stream>>>(agg[dt], bb, (long)Nd * 64);
        }
        for (int t = 0; t < 3; ++t)
            layer_update<<<CDIV((long)nn[t] * 64, 256), 256, 0, stream>>>(h[t], agg[t], (long)nn[t] * 64);
    }

    // ---- classifier: 16x32 strip per wave (NT=2) ----
    {
        int groups = (NP >> 4) * ((32 >> 4) / 2);
        gemm_wmma_packed<2><<<CDIV(groups, 4), 128, 0, stream>>>(h[0], 64, c1Wp, c1b, z, 32,
                                                                 NP, 32, 64);
    }
    cls2k<<<CDIV(NP, 256), 256, 0, stream>>>(z, c2W, c2b, (float*)d_out, NP);    // relu fused
}
